// SDEGenerator_67388036874698
// MI455X (gfx1250) — matrix operand order repaired and checked by reading, not compiled, not run on hardware
//
#include <hip/hip_runtime.h>
#include <hip/hip_bf16.h>

#define BDIM   1024
#define LATD   64
#define HIDD   128
#define NOISED 16
#define OUTD   32
#define INOISE 64
#define TT     128

typedef __attribute__((ext_vector_type(16))) _Float16 v16h;
typedef __attribute__((ext_vector_type(8)))  _Float16 v8h;
typedef __attribute__((ext_vector_type(8)))  float    v8f;

// ---------------------------------------------------------------------------
// WMMA fragment loaders (this file's contiguous k order for A and B, wave32)
// A (16x32 f16): lane holds row M=lane&15; elements 0..15 -> K = half*16 + 0..15
//                (half = lane>>4)
// ---------------------------------------------------------------------------
__device__ __forceinline__ v16h load_a(const _Float16* A, int strideK, int kc, int lane) {
    const int m = lane & 15;
    const int h = (lane >> 4) * 16;
    const _Float16* p = A + m * strideK + kc + h;
    v8h lo = *(const v8h*)(p);
    v8h hi = *(const v8h*)(p + 8);
    v16h r;
#pragma unroll
    for (int e = 0; e < 8; e++) { r[e] = lo[e]; r[e + 8] = hi[e]; }
    return r;
}

// B (32x16 f16): lane holds col N=lane&15; K = (lane>>4)*16 + e, e=0..15
// contiguous -> single 32-byte load from row-major W (N rows x K cols).
__device__ __forceinline__ v16h load_b(const _Float16* W, int K, int ntile, int kc, int lane) {
    const int n = lane & 15;
    const int h = (lane >> 4) * 16;
    const _Float16* p = W + (size_t)(ntile * 16 + n) * K + kc + h;
    return *(const v16h*)p;
}

// B fragment from an LDS-resident 16x128 tile (row-major, stride 128)
__device__ __forceinline__ v16h load_b_lds(const _Float16* buf, int kc, int lane) {
    const int n = lane & 15;
    const int h = (lane >> 4) * 16;
    return *(const v16h*)(buf + n * 128 + kc + h);
}

// One 16x16 output tile: acc = A(16xK) * W^T(Kx16), K multiple of 32, strideA==K
template<int K>
__device__ __forceinline__ v8f wmma_tile(const _Float16* A, const _Float16* W, int ntile, int lane) {
    v8f acc = {0.f, 0.f, 0.f, 0.f, 0.f, 0.f, 0.f, 0.f};
#pragma unroll
    for (int kc = 0; kc < K; kc += 32) {
        v16h a = load_a(A, K, kc, lane);
        v16h b = load_b(W, K, ntile, kc, lane);
        acc = __builtin_amdgcn_wmma_f32_16x16x32_f16(false, a, false, b,
                                                     (short)0, acc, false, false);
    }
    return acc;
}

__device__ __forceinline__ float lipswish(float x) {
    return 0.909f * x / (1.0f + __expf(-x));
}

// ---------------------------------------------------------------------------
// Async Global -> LDS staging (CDNA5 ISA, ASYNCcnt).
// Streams one 16x128 f16 weight tile (4 KB) with 8 x b128 ops (16 B / lane).
// ---------------------------------------------------------------------------
__device__ __forceinline__ void async_load_tile(const _Float16* Wtile, uint32_t ldsbase,
                                                int lane) {
    uint64_t g = (uint64_t)(uintptr_t)Wtile + (uint32_t)lane * 16u;
    uint32_t l = ldsbase + (uint32_t)lane * 16u;
#pragma unroll
    for (int q = 0; q < 8; q++) {
        asm volatile("global_load_async_to_lds_b128 %0, %1, off"
                     :: "v"(l), "v"(g) : "memory");
        g += 512u;
        l += 512u;
    }
}

// ---------------------------------------------------------------------------
// Prep: weight-norm (if g != null), f16 convert, zero-pad K -> Kpad.
// One block per output row.
// ---------------------------------------------------------------------------
__global__ __launch_bounds__(128) void prep_weights(const float* __restrict__ v,
                                                    const float* __restrict__ g,
                                                    _Float16* __restrict__ dst,
                                                    int Kin, int Kpad) {
    __shared__ float red[128];
    const int n = blockIdx.x;
    const int tid = threadIdx.x;
    float ss = 0.f;
    for (int k = tid; k < Kin; k += 128) { float x = v[(size_t)n * Kin + k]; ss += x * x; }
    red[tid] = ss;
    __syncthreads();
    for (int off = 64; off > 0; off >>= 1) {
        if (tid < off) red[tid] += red[tid + off];
        __syncthreads();
    }
    float scale = 1.0f;
    if (g) scale = g[n] / sqrtf(red[0]);
    for (int k = tid; k < Kpad; k += 128)
        dst[(size_t)n * Kpad + k] = (_Float16)((k < Kin) ? v[(size_t)n * Kin + k] * scale : 0.0f);
}

// ---------------------------------------------------------------------------
// Scan kernel helpers (one 16-row batch tile, 8 waves, 256 threads)
// ---------------------------------------------------------------------------
__device__ __forceinline__ void build_tz(_Float16* s_tz, const float* s_z, float t, int tid) {
#pragma unroll
    for (int jj = 0; jj < 6; jj++) {            // 16*96 / 256 = 6, straight-line
        int j = tid + jj * 256;
        int c = j % 96;
        float v = (c == 0) ? t : ((c <= 64) ? s_z[(j / 96) * LATD + (c - 1)] : 0.0f);
        s_tz[j] = (_Float16)v;
    }
    __syncthreads();
}

// Two hidden layers with lipswish: tz(16x96) -> hA(16x128) -> hB(16x128)
__device__ __forceinline__ void mlp12(const _Float16* s_tz, _Float16* s_hA, _Float16* s_hB,
                                      const _Float16* W1, const float* b1,
                                      const _Float16* W2, const float* b2,
                                      int wave, int lane) {
    v8f a1 = wmma_tile<96>(s_tz, W1, wave, lane);
    {
        int n = wave * 16 + (lane & 15), mh = (lane >> 4) * 8;
        float bb = b1[n];
#pragma unroll
        for (int r = 0; r < 8; r++) s_hA[(mh + r) * HIDD + n] = (_Float16)lipswish(a1[r] + bb);
    }
    __syncthreads();
    v8f a2 = wmma_tile<128>(s_hA, W2, wave, lane);
    {
        int n = wave * 16 + (lane & 15), mh = (lane >> 4) * 8;
        float bb = b2[n];
#pragma unroll
        for (int r = 0; r < 8; r++) s_hB[(mh + r) * HIDD + n] = (_Float16)lipswish(a2[r] + bb);
    }
    __syncthreads();
}

// Drift output layer: hB(16x128) -> f(16x64) f32
__device__ __forceinline__ void drift3(const _Float16* s_hB, const _Float16* W3,
                                       const float* b3, float* fdst, int wave, int lane) {
    if (wave < 4) {
        v8f a3 = wmma_tile<128>(s_hB, W3, wave, lane);
        int n = wave * 16 + (lane & 15), mh = (lane >> 4) * 8;
        float bb = b3[n];
#pragma unroll
        for (int r = 0; r < 8; r++) fdst[(mh + r) * LATD + n] = a3[r] + bb;
    }
    __syncthreads();
}

// Diffusion output layer: hB(16x128) -> g(16x1024) f16.
// Each wave owns 8 column tiles of gW3; the 4 KB weight tile for step t8 is
// async-streamed into a per-wave LDS double buffer, overlapped with the WMMAs
// of the previous tile. In-order ASYNCcnt retirement => waiting to <=8 with
// 16 outstanding guarantees the older tile is resident.
__device__ __forceinline__ void diff3_async(const _Float16* s_hB, const _Float16* gW3,
                                            const float* b3, _Float16* gdst,
                                            _Float16* wbuf /* per-wave 2 x 2048 halfs */,
                                            int wave, int lane) {
    const uint32_t lds0 = (uint32_t)(uintptr_t)wbuf;
    const uint32_t lds1 = lds0 + 4096u;

    async_load_tile(gW3 + (size_t)wave * 16 * 128, lds0, lane);
    async_load_tile(gW3 + (size_t)(wave + 8) * 16 * 128, lds1, lane);

#pragma unroll 1
    for (int t8 = 0; t8 < 8; t8++) {
        if (t8 == 7) asm volatile("s_wait_asynccnt 0x0" ::: "memory");
        else         asm volatile("s_wait_asynccnt 0x8" ::: "memory");

        const _Float16* buf = (t8 & 1) ? (wbuf + 2048) : wbuf;
        v8f acc = {0.f, 0.f, 0.f, 0.f, 0.f, 0.f, 0.f, 0.f};
#pragma unroll
        for (int kc = 0; kc < 128; kc += 32) {
            v16h a = load_a(s_hB, 128, kc, lane);
            v16h b = load_b_lds(buf, kc, lane);
            acc = __builtin_amdgcn_wmma_f32_16x16x32_f16(false, a, false, b,
                                                         (short)0, acc, false, false);
        }
        const int nt = wave + t8 * 8;
        if (t8 < 6) {
            // fragment ds_loads of this buffer were consumed by the WMMAs;
            // drain DScnt before the async engine overwrites the buffer.
            asm volatile("s_wait_dscnt 0x0" ::: "memory");
            async_load_tile(gW3 + (size_t)(nt + 16) * 16 * 128,
                            (t8 & 1) ? lds1 : lds0, lane);
        }
        int n = nt * 16 + (lane & 15), mh = (lane >> 4) * 8;
        float bb = b3[n];
#pragma unroll
        for (int r = 0; r < 8; r++) gdst[(mh + r) * (LATD * NOISED) + n] = (_Float16)(acc[r] + bb);
    }
    asm volatile("s_wait_asynccnt 0x0" ::: "memory");
    __syncthreads();
}

// Readout: y -> lipswish(yh @ rW1^T + b) @ rW2^T + b -> out[:, tidx, :]
// (s_yh is maintained incrementally by the y-update / y0 epilogues.)
__device__ __forceinline__ void readout(const _Float16* s_yh, _Float16* s_hA,
                                        const _Float16* rW1, const float* rb1,
                                        const _Float16* rW2, const float* rb2,
                                        float* __restrict__ out, int row0, int tidx,
                                        int wave, int lane) {
    v8f h = wmma_tile<64>(s_yh, rW1, wave, lane);
    {
        int n = wave * 16 + (lane & 15), mh = (lane >> 4) * 8;
        float bb = rb1[n];
#pragma unroll
        for (int r = 0; r < 8; r++) s_hA[(mh + r) * HIDD + n] = (_Float16)lipswish(h[r] + bb);
    }
    __syncthreads();
    if (wave < 2) {
        v8f x = wmma_tile<128>(s_hA, rW2, wave, lane);
        int n = wave * 16 + (lane & 15), mh = (lane >> 4) * 8;
        float bb = rb2[n];
#pragma unroll
        for (int r = 0; r < 8; r++)
            out[(size_t)(row0 + mh + r) * (TT * OUTD) + (size_t)tidx * OUTD + n] = x[r] + bb;
    }
    __syncthreads();
}

// ---------------------------------------------------------------------------
// Persistent-per-batch-tile scan: 64 blocks x 256 threads, each block owns
// 16 batch rows and runs all 127 steps internally. Weights live in L2; the
// big diffusion output weight is async-streamed through LDS double buffers.
// ---------------------------------------------------------------------------
__global__ __launch_bounds__(256) void sde_scan(
    const _Float16* __restrict__ noiseh, const _Float16* __restrict__ embW,
    const float* __restrict__ embb,
    const _Float16* __restrict__ dW1, const float* __restrict__ db1,
    const _Float16* __restrict__ dW2, const float* __restrict__ db2,
    const _Float16* __restrict__ dW3, const float* __restrict__ db3,
    const _Float16* __restrict__ gW1, const float* __restrict__ gb1,
    const _Float16* __restrict__ gW2, const float* __restrict__ gb2,
    const _Float16* __restrict__ gW3, const float* __restrict__ gb3,
    const _Float16* __restrict__ rW1, const float* __restrict__ rb1,
    const _Float16* __restrict__ rW2, const float* __restrict__ rb2,
    const float* __restrict__ ts, const float* __restrict__ bm,
    float* __restrict__ out) {

    __shared__ __align__(32) _Float16 s_tz[16 * 96];            //  3 KB
    __shared__ __align__(32) _Float16 s_hA[16 * HIDD];          //  4 KB
    __shared__ __align__(32) _Float16 s_hB[16 * HIDD];          //  4 KB
    __shared__ __align__(32) _Float16 s_yh[16 * LATD];          //  2 KB
    __shared__ __align__(32) _Float16 s_g[16 * LATD * NOISED];  // 32 KB
    __shared__ __align__(32) _Float16 s_wbuf[8][2][2048];       // 64 KB async staging
    __shared__ float s_y[16 * LATD];                            //  4 KB
    __shared__ float s_z[16 * LATD];                            //  4 KB
    __shared__ float s_f[16 * LATD];                            //  4 KB
    __shared__ float s_dw[16 * NOISED];                         //  1 KB

    const int tid = threadIdx.x;
    const int wave = tid >> 5;
    const int lane = tid & 31;
    const int row0 = blockIdx.x * 16;
    _Float16* mybuf = &s_wbuf[wave][0][0];

    // ---- y0 = init_noise @ embW^T + emb_b ----
#pragma unroll
    for (int jj = 0; jj < 4; jj++) {            // 16*64 / 256 = 4
        int j = tid + jj * 256;
        s_yh[j] = noiseh[(size_t)(row0 + (j >> 6)) * INOISE + (j & 63)];
    }
    __syncthreads();
    if (wave < 4) {
        v8f acc = wmma_tile<64>(s_yh, embW, wave, lane);
        int n = wave * 16 + (lane & 15), mh = (lane >> 4) * 8;
        float bb = embb[n];
#pragma unroll
        for (int r = 0; r < 8; r++) {
            float v = acc[r] + bb;
            s_y[(mh + r) * LATD + n] = v;
            s_z[(mh + r) * LATD + n] = v;
        }
    }
    __syncthreads();
    // f16 copy of y for the readout A-fragments (overwrites the noise tile)
#pragma unroll
    for (int jj = 0; jj < 4; jj++) {
        int j = tid + jj * 256;
        s_yh[j] = (_Float16)s_y[j];
    }
    __syncthreads();

    // t = 0 readout, then f0 / g0
    readout(s_yh, s_hA, rW1, rb1, rW2, rb2, out, row0, 0, wave, lane);
    build_tz(s_tz, s_z, ts[0], tid);
    mlp12(s_tz, s_hA, s_hB, dW1, db1, dW2, db2, wave, lane);
    drift3(s_hB, dW3, db3, s_f, wave, lane);
    mlp12(s_tz, s_hA, s_hB, gW1, gb1, gW2, gb2, wave, lane);
    diff3_async(s_hB, gW3, gb3, s_g, mybuf, wave, lane);

    // per-thread register carry: this thread owns elements idx = tid*4 + j
    float fprev[4];
#pragma unroll
    for (int j = 0; j < 4; j++) fprev[j] = s_f[tid * 4 + j];

#pragma unroll 1
    for (int i = 0; i < TT - 1; i++) {
        float t1  = ts[i + 1];
        float dti = t1 - ts[i];
        float sq  = sqrtf(dti);

        // dW_i = bm_noise[i] * sqrt(dt)   (16 rows x 16 noise -> one per thread)
        s_dw[tid] = bm[(size_t)i * BDIM * NOISED +
                       (size_t)(row0 + (tid >> 4)) * NOISED + (tid & 15)] * sq;
        __syncthreads();

        // z1 = 2y - z + f*dt + g_prev . dW  (keep g_prev.dW in registers)
        float gdp[4];
#pragma unroll
        for (int j = 0; j < 4; j++) {
            int idx = tid * 4 + j;
            int b = idx >> 6, l = idx & 63;
            float gd = 0.f;
#pragma unroll
            for (int n2 = 0; n2 < NOISED; n2++)
                gd += (float)s_g[b * (LATD * NOISED) + l * NOISED + n2] * s_dw[b * NOISED + n2];
            gdp[j] = gd;
            s_z[idx] = 2.f * s_y[idx] - s_z[idx] + fprev[j] * dti + gd;
        }
        __syncthreads();

        build_tz(s_tz, s_z, t1, tid);
        mlp12(s_tz, s_hA, s_hB, dW1, db1, dW2, db2, wave, lane);
        drift3(s_hB, dW3, db3, s_f, wave, lane);               // s_f <- f_{i+1}
        mlp12(s_tz, s_hA, s_hB, gW1, gb1, gW2, gb2, wave, lane);
        diff3_async(s_hB, gW3, gb3, s_g, mybuf, wave, lane);   // s_g <- g_{i+1}

        // y1 = y + 0.5(f+f1)dt + 0.5(g.dW + g1.dW); also refresh f16 copy s_yh
#pragma unroll
        for (int j = 0; j < 4; j++) {
            int idx = tid * 4 + j;
            int b = idx >> 6, l = idx & 63;
            float gd = 0.f;
#pragma unroll
            for (int n2 = 0; n2 < NOISED; n2++)
                gd += (float)s_g[b * (LATD * NOISED) + l * NOISED + n2] * s_dw[b * NOISED + n2];
            float y1 = s_y[idx] + 0.5f * (fprev[j] + s_f[idx]) * dti + 0.5f * (gdp[j] + gd);
            s_y[idx]  = y1;
            s_yh[idx] = (_Float16)y1;
            fprev[j]  = s_f[idx];
        }
        __syncthreads();

        readout(s_yh, s_hA, rW1, rb1, rW2, rb2, out, row0, i + 1, wave, lane);
    }
}

// x_matched[b, s, :] = x_full[b, s*multiplier, :]
__global__ __launch_bounds__(256) void gather_matched(const float* __restrict__ xfull,
                                                      float* __restrict__ xm,
                                                      const int* __restrict__ multp,
                                                      int stepsN, int total) {
    int idx = blockIdx.x * blockDim.x + threadIdx.x;
    if (idx >= total) return;
    int mult = multp[0];
    int o = idx % OUTD;
    int s = (idx / OUTD) % stepsN;
    int b = idx / (OUTD * stepsN);
    xm[idx] = xfull[(size_t)b * TT * OUTD + (size_t)(s * mult) * OUTD + o];
}

// ---------------------------------------------------------------------------
extern "C" void kernel_launch(void* const* d_in, const int* in_sizes, int n_in,
                              void* d_out, int out_size, void* d_ws, size_t ws_size,
                              hipStream_t stream) {
    (void)in_sizes; (void)n_in; (void)ws_size;

    const float* init_noise = (const float*)d_in[0];
    const float* bm         = (const float*)d_in[1];
    const float* ts         = (const float*)d_in[2];
    const int*   mult       = (const int*)d_in[4];
    const float* emb_W = (const float*)d_in[5];
    const float* emb_b = (const float*)d_in[6];
    const float* dr_v1 = (const float*)d_in[7],  *dr_g1 = (const float*)d_in[8],  *dr_b1 = (const float*)d_in[9];
    const float* dr_v2 = (const float*)d_in[10], *dr_g2 = (const float*)d_in[11], *dr_b2 = (const float*)d_in[12];
    const float* dr_v3 = (const float*)d_in[13], *dr_g3 = (const float*)d_in[14], *dr_b3 = (const float*)d_in[15];
    const float* di_v1 = (const float*)d_in[16], *di_g1 = (const float*)d_in[17], *di_b1 = (const float*)d_in[18];
    const float* di_v2 = (const float*)d_in[19], *di_g2 = (const float*)d_in[20], *di_b2 = (const float*)d_in[21];
    const float* di_v3 = (const float*)d_in[22], *di_g3 = (const float*)d_in[23], *di_b3 = (const float*)d_in[24];
    const float* ro_W1 = (const float*)d_in[25], *ro_b1 = (const float*)d_in[26];
    const float* ro_W2 = (const float*)d_in[27], *ro_b2 = (const float*)d_in[28];

    // Workspace layout (f16 elements, all offsets 32B-aligned)
    _Float16* w     = (_Float16*)d_ws;
    _Float16* embWh = w + 0;        //  64 x  64
    _Float16* dW1   = w + 4096;     // 128 x  96 (K 65 -> 96)
    _Float16* dW2   = w + 16384;    // 128 x 128
    _Float16* dW3   = w + 32768;    //  64 x 128
    _Float16* gW1   = w + 40960;    // 128 x  96
    _Float16* gW2   = w + 53248;    // 128 x 128
    _Float16* gW3   = w + 69632;    // 1024 x 128
    _Float16* rW1   = w + 200704;   // 128 x  64
    _Float16* rW2   = w + 208896;   //  32 x 128
    _Float16* nzh   = w + 212992;   // 1024 x 64 (init_noise)

    prep_weights<<<64,   128, 0, stream>>>(emb_W, nullptr, embWh, 64, 64);
    prep_weights<<<128,  128, 0, stream>>>(dr_v1, dr_g1, dW1, 65, 96);
    prep_weights<<<128,  128, 0, stream>>>(dr_v2, dr_g2, dW2, 128, 128);
    prep_weights<<<64,   128, 0, stream>>>(dr_v3, dr_g3, dW3, 128, 128);
    prep_weights<<<128,  128, 0, stream>>>(di_v1, di_g1, gW1, 65, 96);
    prep_weights<<<128,  128, 0, stream>>>(di_v2, di_g2, gW2, 128, 128);
    prep_weights<<<1024, 128, 0, stream>>>(di_v3, di_g3, gW3, 128, 128);
    prep_weights<<<128,  128, 0, stream>>>(ro_W1, nullptr, rW1, 64, 64);
    prep_weights<<<32,   128, 0, stream>>>(ro_W2, nullptr, rW2, 128, 128);
    prep_weights<<<1024, 128, 0, stream>>>(init_noise, nullptr, nzh, 64, 64);

    float* xfull = (float*)d_out;
    sde_scan<<<BDIM / 16, 256, 0, stream>>>(nzh, embWh, emb_b,
                                            dW1, dr_b1, dW2, dr_b2, dW3, dr_b3,
                                            gW1, di_b1, gW2, di_b2, gW3, di_b3,
                                            rW1, ro_b1, rW2, ro_b2,
                                            ts, bm, xfull);

    int n_match = out_size - BDIM * TT * OUTD;
    if (n_match > 0) {
        int stepsN = n_match / (BDIM * OUTD);
        gather_matched<<<(n_match + 255) / 256, 256, 0, stream>>>(
            xfull, xfull + (size_t)BDIM * TT * OUTD, mult, stepsN, n_match);
    }
}
